// Net_9277129360139
// MI455X (gfx1250) — matrix operand order repaired and checked by reading, not compiled, not run on hardware
//
#include <hip/hip_runtime.h>
#include <hip/hip_bf16.h>
#include <math.h>

#define Nn 16384
#define Ee 65536
#define Pp 32768

typedef __attribute__((ext_vector_type(16))) _Float16 v16h;
typedef __attribute__((ext_vector_type(8)))  _Float16 v8h;
typedef __attribute__((ext_vector_type(4)))  _Float16 h4v;
typedef __attribute__((ext_vector_type(8)))  float    v8f;
typedef __attribute__((ext_vector_type(4)))  float    v4f;

static __device__ __forceinline__ v8f wmma_f16(v16h a, v16h b, v8f c) {
  return __builtin_amdgcn_wmma_f32_16x16x32_f16(false, a, false, b, (short)0, c, false, false);
}

// ---- fragment loaders -------------------------------------------------------
// A fragment (16x32 f16) from f16 row pointer (row = lane%16):
// elem j <-> K = kbase + hid*8 + j ; elem j+8 <-> K = kbase + 16 + hid*8 + j
static __device__ __forceinline__ v16h load_a16(const _Float16* rowp, int kbase) {
  int hid = (threadIdx.x & 31) >> 4;
  v8h lo = *(const v8h*)(rowp + kbase + hid * 8);
  v8h hi = *(const v8h*)(rowp + kbase + 16 + hid * 8);
  return __builtin_shufflevector(lo, hi, 0, 1, 2, 3, 4, 5, 6, 7,
                                 8, 9, 10, 11, 12, 13, 14, 15);
}

// A fragment from f32 row pointer (inline convert; used for small node kernels)
static __device__ __forceinline__ v16h load_a32(const float* rowp, int kbase) {
  int hid = (threadIdx.x & 31) >> 4;
  v16h a;
#pragma unroll
  for (int j = 0; j < 8; ++j) {
    a[j]     = (_Float16)rowp[kbase + hid * 8 + j];
    a[j + 8] = (_Float16)rowp[kbase + 16 + hid * 8 + j];
  }
  return a;
}

// B fragment (32x16 f16) for C = A @ W^T, W f16 row-major [*, ldw]:
// col n = lane%16 ; elem j <-> k = kbase + hid*8 + j ; elem j+8 <-> k = kbase + 16 + hid*8 + j
static __device__ __forceinline__ v16h load_b16(const _Float16* W, int ldw, int n0, int kbase) {
  int lane = threadIdx.x & 31, r = lane & 15, hid = lane >> 4;
  const _Float16* p = W + (size_t)(n0 + r) * ldw + kbase + hid * 8;
  v8h lo = *(const v8h*)(p);
  v8h hi = *(const v8h*)(p + 16);
  return __builtin_shufflevector(lo, hi, 0, 1, 2, 3, 4, 5, 6, 7,
                                 8, 9, 10, 11, 12, 13, 14, 15);
}

static __device__ __forceinline__ float sigmoidf_(float x) { return 1.f / (1.f + __expf(-x)); }

// ---- utility kernels --------------------------------------------------------
__global__ void k_fill(float* p, size_t n, float v) {
  size_t i = (size_t)blockIdx.x * blockDim.x + threadIdx.x;
  size_t st = (size_t)gridDim.x * blockDim.x;
  for (; i < n; i += st) p[i] = v;
}

__global__ void k_cvt(const float* __restrict__ s, _Float16* __restrict__ d, size_t n) {
  size_t i = (size_t)blockIdx.x * blockDim.x + threadIdx.x;
  size_t st = (size_t)gridDim.x * blockDim.x;
  for (; i < n; i += st) d[i] = (_Float16)s[i];
}

// lin0_w [64,16] f32 -> f16 padded to [64,32] (K padded with zeros)
__global__ void k_cvt_pad_lin0(const float* __restrict__ s, _Float16* __restrict__ d) {
  int i = blockIdx.x * blockDim.x + threadIdx.x;  // 64*32
  if (i >= 64 * 32) return;
  int n = i >> 5, k = i & 31;
  d[i] = (k < 16) ? (_Float16)s[n * 16 + k] : (_Float16)0.f;
}

// ---- h = relu(x @ lin0_w^T + b) ; x:[N,16] (K padded 16->32) ----------------
__global__ void k_lin0(const float* __restrict__ x, const _Float16* __restrict__ w32,
                       const float* __restrict__ b, float* __restrict__ h) {
  int wid = (blockIdx.x * blockDim.x + threadIdx.x) >> 5;
  int lane = threadIdx.x & 31, r = lane & 15, hid = lane >> 4;
  int row0 = wid * 16;
  const float* rowp = x + (size_t)(row0 + r) * 16;
  v16h a;
#pragma unroll
  for (int j = 0; j < 8; ++j) { a[j] = (_Float16)rowp[hid * 8 + j]; a[j + 8] = (_Float16)0.f; }
#pragma unroll
  for (int t = 0; t < 4; ++t) {
    int n0 = 16 * t;
    v8f acc = {};
    acc = wmma_f16(a, load_b16(w32, 32, n0, 0), acc);
    float bias = b[n0 + r];
#pragma unroll
    for (int i = 0; i < 8; ++i) {
      int orow = row0 + i + 8 * hid;
      h[(size_t)orow * 64 + n0 + r] = fmaxf(acc[i] + bias, 0.f);
    }
  }
}

// ---- e1 = relu(LN(edge_attr @ w1^T + b1)) -> f16 ; 1 block / edge -----------
__global__ void k_mlp1(const float* __restrict__ ea, const float* __restrict__ w1,
                       const float* __restrict__ b1, const float* __restrict__ g1,
                       const float* __restrict__ be1, _Float16* __restrict__ e1) {
  int e = blockIdx.x, f = threadIdx.x;
  const float* a = ea + (size_t)e * 4;
  const float* wr = w1 + (size_t)f * 4;
  float v = b1[f] + a[0] * wr[0] + a[1] * wr[1] + a[2] * wr[2] + a[3] * wr[3];
  __shared__ float s1[128], s2[128];
  s1[f] = v; s2[f] = v * v; __syncthreads();
  for (int s = 64; s > 0; s >>= 1) {
    if (f < s) { s1[f] += s1[f + s]; s2[f] += s2[f + s]; }
    __syncthreads();
  }
  float m = s1[0] * (1.f / 128.f);
  float var = s2[0] * (1.f / 128.f) - m * m;
  e1[(size_t)e * 128 + f] =
      (_Float16)fmaxf((v - m) * rsqrtf(var + 1e-5f) * g1[f] + be1[f], 0.f);
}

// ---- fused: T[e,4096] = LN(e1[e,:] @ w2^T + b2)*g2+be2  (f16 out) -----------
// One block = 16 edge rows. 16 waves, each owns 16 column tiles kept entirely
// in registers (v8f x16); LN stats combined through 32 floats of LDS.
__global__ void k_wgen(const _Float16* __restrict__ e1, const _Float16* __restrict__ w2,
                       const float* __restrict__ b2, const float* __restrict__ g2,
                       const float* __restrict__ be2, _Float16* __restrict__ T) {
  __shared__ float ss[16], sq[16];
  int erow0 = blockIdx.x * 16;
  int w = threadIdx.x >> 5;
  int lane = threadIdx.x & 31, r = lane & 15, hid = lane >> 4;
  if (threadIdx.x < 16) { ss[threadIdx.x] = 0.f; sq[threadIdx.x] = 0.f; }
  __syncthreads();

  const _Float16* rowp = e1 + (size_t)(erow0 + r) * 128;
  v16h A[4];
#pragma unroll
  for (int kc = 0; kc < 4; ++kc) A[kc] = load_a16(rowp, kc * 32);

  v8f acc[16];
#pragma unroll
  for (int tt = 0; tt < 16; ++tt) {
    int n0 = (w * 16 + tt) * 16;
    v8f a = {};
#pragma unroll
    for (int kc = 0; kc < 4; ++kc) a = wmma_f16(A[kc], load_b16(w2, 128, n0, kc * 32), a);
    float bb = b2[n0 + r];
#pragma unroll
    for (int i = 0; i < 8; ++i) a[i] += bb;
    acc[tt] = a;
  }

  // per-row partial stats over this wave's 256 columns
#pragma unroll
  for (int i = 0; i < 8; ++i) {
    float s = 0.f, q = 0.f;
#pragma unroll
    for (int tt = 0; tt < 16; ++tt) { float v = acc[tt][i]; s += v; q += v * v; }
    for (int m = 1; m < 16; m <<= 1) { s += __shfl_xor(s, m, 32); q += __shfl_xor(q, m, 32); }
    if (r == 0) { atomicAdd(&ss[i + 8 * hid], s); atomicAdd(&sq[i + 8 * hid], q); }
  }
  __syncthreads();

  float mn[8], iv[8];
#pragma unroll
  for (int i = 0; i < 8; ++i) {
    float m_ = ss[i + 8 * hid] * (1.f / 4096.f);
    iv[i] = rsqrtf(sq[i + 8 * hid] * (1.f / 4096.f) - m_ * m_ + 1e-5f);
    mn[i] = m_;
  }
#pragma unroll
  for (int tt = 0; tt < 16; ++tt) {
    int n0 = (w * 16 + tt) * 16;
    float gg = g2[n0 + r], bb = be2[n0 + r];
#pragma unroll
    for (int i = 0; i < 8; ++i)
      T[(size_t)(erow0 + i + 8 * hid) * 4096 + n0 + r] =
          (_Float16)((acc[tt][i] - mn[i]) * iv[i] * gg + bb);
  }
}

__global__ void k_deg(const int* __restrict__ ei, float* __restrict__ deg) {
  int e = blockIdx.x * blockDim.x + threadIdx.x;
  if (e < Ee) atomicAdd(&deg[ei[Ee + e]], 1.f);
}
__global__ void k_inv(const float* __restrict__ deg, float* __restrict__ inv) {
  int i = blockIdx.x * blockDim.x + threadIdx.x;
  if (i < Nn) inv[i] = 1.f / fmaxf(deg[i], 1.f);
}

// ---- msg[e,:] = h[src[e],:] @ W_e ; atomic scatter into agg[dst] ------------
// One wave per edge; low/high half-waves consume even/odd rows of W_e so each
// iteration is one b64 load per lane (two 128B rows per wave-iteration).
__global__ void k_msg(const int* __restrict__ ei, const float* __restrict__ h,
                      const _Float16* __restrict__ T, float* __restrict__ agg) {
  int e = (blockIdx.x * blockDim.x + threadIdx.x) >> 5;
  int lane = threadIdx.x & 31, r = lane & 15, hpart = lane >> 4;
  int s = ei[e], d = ei[Ee + e];
  const float* hp = h + (size_t)s * 64;
  float h0 = hp[lane], h1 = hp[32 + lane];
  const _Float16* wp = T + (size_t)e * 4096 + 4 * lane;
  float a0 = 0.f, a1 = 0.f, a2 = 0.f, a3 = 0.f;
#pragma unroll 8
  for (int p = 0; p < 32; ++p) {
    // row index this lane consumes: d_row = 2p + hpart
    float hd = __shfl(p < 16 ? h0 : h1, (2 * p + hpart) & 31, 32);
    h4v wv = *(const h4v*)(wp + (size_t)p * 128);
    a0 = fmaf(hd, (float)wv[0], a0);
    a1 = fmaf(hd, (float)wv[1], a1);
    a2 = fmaf(hd, (float)wv[2], a2);
    a3 = fmaf(hd, (float)wv[3], a3);
  }
  a0 += __shfl_xor(a0, 16, 32);
  a1 += __shfl_xor(a1, 16, 32);
  a2 += __shfl_xor(a2, 16, 32);
  a3 += __shfl_xor(a3, 16, 32);
  float* op = agg + (size_t)d * 64 + 4 * r + 2 * hpart;
  atomicAdd(op, hpart ? a2 : a0);
  atomicAdd(op + 1, hpart ? a3 : a1);
}

// ---- fused GRU cell: m = relu(agg*inv_deg + cb); h = GRU(m, h) --------------
__global__ void k_gru(const float* __restrict__ agg, const float* __restrict__ inv,
                      const float* __restrict__ cb, const _Float16* __restrict__ wih,
                      const _Float16* __restrict__ whh, const float* __restrict__ bih,
                      const float* __restrict__ bhh, float* __restrict__ h) {
  int wid = (blockIdx.x * blockDim.x + threadIdx.x) >> 5;
  int lane = threadIdx.x & 31, r = lane & 15, hid = lane >> 4;
  int row0 = wid * 16, row = row0 + r;
  float invd = inv[row];
  const float* ap = agg + (size_t)row * 64;
  const float* hp = h + (size_t)row * 64;
  v16h mA[2], hA[2];
#pragma unroll
  for (int kc = 0; kc < 2; ++kc) {
#pragma unroll
    for (int j = 0; j < 8; ++j) {
      int k1 = kc * 32 + hid * 8 + j, k2 = k1 + 16;
      mA[kc][j]     = (_Float16)fmaxf(fmaf(ap[k1], invd, cb[k1]), 0.f);
      mA[kc][j + 8] = (_Float16)fmaxf(fmaf(ap[k2], invd, cb[k2]), 0.f);
      hA[kc][j]     = (_Float16)hp[k1];
      hA[kc][j + 8] = (_Float16)hp[k2];
    }
  }
  v8f gi[12], gh[12];
#pragma unroll
  for (int t = 0; t < 12; ++t) {
    int n0 = 16 * t;
    v8f ai = {}, ah = {};
#pragma unroll
    for (int kc = 0; kc < 2; ++kc) {
      ai = wmma_f16(mA[kc], load_b16(wih, 64, n0, kc * 32), ai);
      ah = wmma_f16(hA[kc], load_b16(whh, 64, n0, kc * 32), ah);
    }
    float bi = bih[n0 + r], bh = bhh[n0 + r];
#pragma unroll
    for (int i = 0; i < 8; ++i) { ai[i] += bi; ah[i] += bh; }
    gi[t] = ai; gh[t] = ah;
  }
#pragma unroll
  for (int j = 0; j < 4; ++j) {
#pragma unroll
    for (int i = 0; i < 8; ++i) {
      int orow = row0 + i + 8 * hid, f = 16 * j + r;
      float rg = sigmoidf_(gi[j][i] + gh[j][i]);
      float z  = sigmoidf_(gi[4 + j][i] + gh[4 + j][i]);
      float nn = tanhf(gi[8 + j][i] + rg * gh[8 + j][i]);
      size_t off = (size_t)orow * 64 + f;
      float ho = h[off];
      h[off] = (1.f - z) * nn + z * ho;
    }
  }
}

// ---- fused Set2Set LSTM step ------------------------------------------------
__global__ void k_lstm(const float* __restrict__ h, float* __restrict__ hs,
                       float* __restrict__ cs, const _Float16* __restrict__ wih,
                       const _Float16* __restrict__ whh, const float* __restrict__ bih,
                       const float* __restrict__ bhh, int first) {
  int wid = (blockIdx.x * blockDim.x + threadIdx.x) >> 5;
  int lane = threadIdx.x & 31, r = lane & 15, hid = lane >> 4;
  int row0 = wid * 16, row = row0 + r;
  const float* hsp = hs + (size_t)row * 64;
  const float* hp  = h + (size_t)row * 64;
  v16h qA[4];
  qA[0] = load_a32(hsp, 0);
  qA[1] = load_a32(hsp, 32);
  if (first) {
    v16h z;
#pragma unroll
    for (int j = 0; j < 16; ++j) z[j] = (_Float16)0.f;
    qA[2] = z; qA[3] = z;
  } else {
    qA[2] = load_a32(hp, 0);
    qA[3] = load_a32(hp, 32);
  }
  v8f g[16];
#pragma unroll
  for (int t = 0; t < 16; ++t) {
    int n0 = 16 * t;
    v8f acc = {};
#pragma unroll
    for (int kc = 0; kc < 4; ++kc) acc = wmma_f16(qA[kc], load_b16(wih, 128, n0, kc * 32), acc);
#pragma unroll
    for (int kc = 0; kc < 2; ++kc) acc = wmma_f16(qA[kc], load_b16(whh, 64, n0, kc * 32), acc);
    float bb = bih[n0 + r] + bhh[n0 + r];
#pragma unroll
    for (int i = 0; i < 8; ++i) acc[i] += bb;
    g[t] = acc;
  }
#pragma unroll
  for (int j = 0; j < 4; ++j) {
#pragma unroll
    for (int i = 0; i < 8; ++i) {
      int orow = row0 + i + 8 * hid, f = 16 * j + r;
      float ii = sigmoidf_(g[j][i]);
      float ff = sigmoidf_(g[4 + j][i]);
      float gg = tanhf(g[8 + j][i]);
      float oo = sigmoidf_(g[12 + j][i]);
      size_t off = (size_t)orow * 64 + f;
      float cn = ff * cs[off] + ii * gg;
      cs[off] = cn;
      hs[off] = oo * tanhf(cn);
    }
  }
}

// ---- pair head: p = relu(LN(feat @ lin1_w^T + b)) ---------------------------
// feat = [out[a0], out[a1], hs[a0], out[a0], hs[a1], out[a1]]  (q_star = [hs,out])
__global__ void k_pair(const float* __restrict__ h, const float* __restrict__ hs,
                       const int* __restrict__ ti, const _Float16* __restrict__ w,
                       const float* __restrict__ b, const float* __restrict__ g,
                       const float* __restrict__ be, float* __restrict__ pbuf) {
  int wid = (blockIdx.x * blockDim.x + threadIdx.x) >> 5;
  int lane = threadIdx.x & 31, r = lane & 15, hid = lane >> 4;
  int row0 = wid * 16, row = row0 + r;
  int a0 = ti[row], a1 = ti[Pp + row];
  const float* segp[6];
  segp[0] = h + (size_t)a0 * 64;
  segp[1] = h + (size_t)a1 * 64;
  segp[2] = hs + (size_t)a0 * 64;
  segp[3] = segp[0];
  segp[4] = hs + (size_t)a1 * 64;
  segp[5] = segp[1];
  v16h A[12];
#pragma unroll
  for (int kc = 0; kc < 12; ++kc) A[kc] = load_a32(segp[kc >> 1], (kc & 1) * 32);
  v8f acc[4];
#pragma unroll
  for (int t = 0; t < 4; ++t) {
    int n0 = 16 * t;
    v8f a = {};
#pragma unroll
    for (int kc = 0; kc < 12; ++kc) a = wmma_f16(A[kc], load_b16(w, 384, n0, kc * 32), a);
    float bias = b[n0 + r];
#pragma unroll
    for (int i = 0; i < 8; ++i) a[i] += bias;
    acc[t] = a;
  }
#pragma unroll
  for (int i = 0; i < 8; ++i) {
    float s = 0.f, sq = 0.f;
#pragma unroll
    for (int t = 0; t < 4; ++t) { float v = acc[t][i]; s += v; sq += v * v; }
    for (int m = 1; m < 16; m <<= 1) { s += __shfl_xor(s, m, 32); sq += __shfl_xor(sq, m, 32); }
    float mn = s * (1.f / 64.f);
    float iv = rsqrtf(sq * (1.f / 64.f) - mn * mn + 1e-5f);
    int orow = row0 + i + 8 * hid;
#pragma unroll
    for (int t = 0; t < 4; ++t) {
      int c = 16 * t + r;
      pbuf[(size_t)orow * 64 + c] = fmaxf((acc[t][i] - mn) * iv * g[c] + be[c], 0.f);
    }
  }
}

// ---- out[p] = (pbuf[p,:] @ lin2_w^T + b)[class[p]] --------------------------
__global__ void k_pred(const float* __restrict__ pbuf, const float* __restrict__ w2,
                       const float* __restrict__ b2, const int* __restrict__ cls,
                       float* __restrict__ out) {
  int t = blockIdx.x * blockDim.x + threadIdx.x;
  if (t >= Pp) return;
  int c = cls[t];
  const v4f* pr = (const v4f*)(pbuf + (size_t)t * 64);
  const v4f* wr = (const v4f*)(w2 + (size_t)c * 64);
  float s = b2[c];
#pragma unroll
  for (int f = 0; f < 16; ++f) {
    v4f p = pr[f], ww = wr[f];
    s = fmaf(p[0], ww[0], s);
    s = fmaf(p[1], ww[1], s);
    s = fmaf(p[2], ww[2], s);
    s = fmaf(p[3], ww[3], s);
  }
  out[t] = s;
}

extern "C" void kernel_launch(void* const* d_in, const int* in_sizes, int n_in,
                              void* d_out, int out_size, void* d_ws, size_t ws_size,
                              hipStream_t stream) {
  const float* x    = (const float*)d_in[0];
  const float* ea   = (const float*)d_in[1];
  const float* l0w  = (const float*)d_in[2];
  const float* l0b  = (const float*)d_in[3];
  const float* w1   = (const float*)d_in[4];
  const float* b1   = (const float*)d_in[5];
  const float* g1   = (const float*)d_in[6];
  const float* be1  = (const float*)d_in[7];
  const float* w2   = (const float*)d_in[8];
  const float* b2   = (const float*)d_in[9];
  const float* g2   = (const float*)d_in[10];
  const float* be2  = (const float*)d_in[11];
  const float* cb   = (const float*)d_in[12];
  const float* gwih = (const float*)d_in[13];
  const float* gwhh = (const float*)d_in[14];
  const float* gbih = (const float*)d_in[15];
  const float* gbhh = (const float*)d_in[16];
  const float* lwih = (const float*)d_in[17];
  const float* lwhh = (const float*)d_in[18];
  const float* lbih = (const float*)d_in[19];
  const float* lbhh = (const float*)d_in[20];
  const float* l1w  = (const float*)d_in[21];
  const float* l1b  = (const float*)d_in[22];
  const float* l1g  = (const float*)d_in[23];
  const float* l1be = (const float*)d_in[24];
  const float* l2w  = (const float*)d_in[25];
  const float* l2b  = (const float*)d_in[26];
  const int*   ei   = (const int*)d_in[27];
  const int*   ti   = (const int*)d_in[28];
  const int*   tcls = (const int*)d_in[29];
  float* out = (float*)d_out;
  (void)in_sizes; (void)n_in; (void)out_size; (void)ws_size;

  char* wsb = (char*)d_ws;
  size_t off = 0;
  auto alloc = [&](size_t bytes) -> void* {
    void* p = wsb + off;
    off = (off + bytes + 255) & ~(size_t)255;
    return p;
  };
  float*    h     = (float*)alloc((size_t)Nn * 64 * 4);
  _Float16* e1    = (_Float16*)alloc((size_t)Ee * 128 * 2);
  _Float16* T     = (_Float16*)alloc((size_t)Ee * 4096 * 2);
  float*    agg   = (float*)alloc((size_t)Nn * 64 * 4);
  float*    deg   = (float*)alloc((size_t)Nn * 4);
  float*    invd  = (float*)alloc((size_t)Nn * 4);
  float*    hs    = (float*)alloc((size_t)Nn * 64 * 4);
  float*    cs    = (float*)alloc((size_t)Nn * 64 * 4);
  float*    pbuf  = (float*)alloc((size_t)Pp * 64 * 4);
  _Float16* w2h   = (_Float16*)alloc((size_t)4096 * 128 * 2);
  _Float16* gwihh = (_Float16*)alloc((size_t)192 * 64 * 2);
  _Float16* gwhhh = (_Float16*)alloc((size_t)192 * 64 * 2);
  _Float16* lwihh = (_Float16*)alloc((size_t)256 * 128 * 2);
  _Float16* lwhhh = (_Float16*)alloc((size_t)256 * 64 * 2);
  _Float16* l1wh  = (_Float16*)alloc((size_t)64 * 384 * 2);
  _Float16* l0wh  = (_Float16*)alloc((size_t)64 * 32 * 2);

  // one-time f16 weight conversion (all tiny; resident in L2 afterwards)
  k_cvt<<<512, 256, 0, stream>>>(w2, w2h, (size_t)4096 * 128);
  k_cvt<<<48, 256, 0, stream>>>(gwih, gwihh, (size_t)192 * 64);
  k_cvt<<<48, 256, 0, stream>>>(gwhh, gwhhh, (size_t)192 * 64);
  k_cvt<<<128, 256, 0, stream>>>(lwih, lwihh, (size_t)256 * 128);
  k_cvt<<<64, 256, 0, stream>>>(lwhh, lwhhh, (size_t)256 * 64);
  k_cvt<<<96, 256, 0, stream>>>(l1w, l1wh, (size_t)64 * 384);
  k_cvt_pad_lin0<<<8, 256, 0, stream>>>(l0w, l0wh);

  k_lin0<<<Nn / 16 / 8, 256, 0, stream>>>(x, l0wh, l0b, h);
  k_mlp1<<<Ee, 128, 0, stream>>>(ea, w1, b1, g1, be1, e1);
  k_wgen<<<Ee / 16, 512, 0, stream>>>(e1, w2h, b2, g2, be2, T);
  k_fill<<<256, 256, 0, stream>>>(deg, (size_t)Nn, 0.f);
  k_deg<<<Ee / 256, 256, 0, stream>>>(ei, deg);
  k_inv<<<Nn / 256, 256, 0, stream>>>(deg, invd);
  for (int s = 0; s < 3; ++s) {
    k_fill<<<1024, 256, 0, stream>>>(agg, (size_t)Nn * 64, 0.f);
    k_msg<<<Ee / 8, 256, 0, stream>>>(ei, h, T, agg);
    k_gru<<<Nn / 16 / 8, 256, 0, stream>>>(agg, invd, cb, gwihh, gwhhh, gbih, gbhh, h);
  }
  k_fill<<<1024, 256, 0, stream>>>(hs, (size_t)Nn * 64, 0.f);
  k_fill<<<1024, 256, 0, stream>>>(cs, (size_t)Nn * 64, 0.f);
  for (int s = 0; s < 3; ++s)
    k_lstm<<<Nn / 16 / 8, 256, 0, stream>>>(h, hs, cs, lwihh, lwhhh, lbih, lbhh, s == 0);
  k_pair<<<Pp / 16 / 8, 256, 0, stream>>>(h, hs, ti, l1wh, l1b, l1g, l1be, pbuf);
  k_pred<<<Pp / 256, 256, 0, stream>>>(pbuf, l2w, l2b, tcls, out);
}
